// Mamba_66279935312355
// MI455X (gfx1250) — compile-verified, matrix operand order checked by reading, not run on hardware
//
#include <hip/hip_runtime.h>

// ---------- constants ----------
#define DM   1024
#define DI   2048
#define DS   16
#define DC   4
#define NB   4
#define SQ   2048
#define NT   (NB*SQ)          // 8192 tokens

// ---------- types ----------
typedef __bf16 vbf8  __attribute__((ext_vector_type(8)));
typedef __bf16 vbf16 __attribute__((ext_vector_type(16)));
typedef float  vf8   __attribute__((ext_vector_type(8)));

__device__ __forceinline__ float bf2f(unsigned short u) {
  union { unsigned u; float f; } v; v.u = ((unsigned)u) << 16; return v.f;
}
__device__ __forceinline__ unsigned short f2bf(float f) {
  union { float f; unsigned u; } v; v.f = f;
  unsigned r = v.u + 0x7FFFu + ((v.u >> 16) & 1u);   // RNE
  return (unsigned short)(r >> 16);
}

// ---------- fp32 -> bf16 convert ----------
__global__ void cvt_bf16_kernel(const float* __restrict__ in,
                                unsigned short* __restrict__ out, int n) {
  int i = blockIdx.x * blockDim.x + threadIdx.x;
  if (i < n) out[i] = f2bf(in[i]);
}

// ---------- WMMA bf16 GEMM:  C[M,N] = A[M,K] * W[N,K]^T ----------
// EPI: 0 = store bf16, 1 = store f32 (1 + exp(v + bias[n])) [= exp(softplus)], 2 = store f32
#define LDT 40   // LDS row stride in halfwords (80B: keeps 16B alignment, spreads banks)

template<int EPI>
__global__ __launch_bounds__(256)
void wmma_gemm_kernel(const unsigned short* __restrict__ A,
                      const unsigned short* __restrict__ W,
                      void* __restrict__ Cout,
                      const float* __restrict__ bias,
                      int M, int N, int K) {
  __shared__ __align__(16) unsigned short sA[128 * LDT];
  __shared__ __align__(16) unsigned short sB[128 * LDT];

  const int tid  = threadIdx.x;
  const int lane = tid & 31;
  const int wv   = tid >> 5;        // 0..7
  const int lh   = lane >> 4;       // half-wave
  const int mr   = lane & 15;       // row/col within fragment
  const int wm   = wv & 1;          // wave M position (2 waves -> 128 rows)
  const int wn   = wv >> 1;         // wave N position (4 waves -> 128 cols)
  const int blockM = blockIdx.y * 128;
  const int blockN = blockIdx.x * 128;

  const vf8 vzero = {0.f,0.f,0.f,0.f,0.f,0.f,0.f,0.f};
  vf8 acc[4][2];
  #pragma unroll
  for (int a = 0; a < 4; ++a)
    #pragma unroll
    for (int b = 0; b < 2; ++b) acc[a][b] = vzero;

  for (int k0 = 0; k0 < K; k0 += 32) {
    // cooperative tile load: 512 chunks of 8 bf16 each per tile
    #pragma unroll
    for (int c = 0; c < 2; ++c) {
      int chunk = tid + c * 256;
      int row = chunk >> 2;
      int col = (chunk & 3) * 8;
      *(vbf8*)(&sA[row * LDT + col]) =
          *(const vbf8*)(A + (size_t)(blockM + row) * K + k0 + col);
      *(vbf8*)(&sB[row * LDT + col]) =
          *(const vbf8*)(W + (size_t)(blockN + row) * K + k0 + col);
    }
    __syncthreads();

    // A fragments: lane<16 holds M=mr; halfword j -> K = 8*lh + j (+8 for j>=8)
    vbf16 af[4], bfr[2];
    #pragma unroll
    for (int fa = 0; fa < 4; ++fa) {
      const unsigned short* p = &sA[(wm * 64 + fa * 16 + mr) * LDT];
      vbf8 lo = *(const vbf8*)(p + 8 * lh);
      vbf8 hi = *(const vbf8*)(p + 16 + 8 * lh);
      #pragma unroll
      for (int j = 0; j < 8; ++j) { af[fa][j] = lo[j]; af[fa][8 + j] = hi[j]; }
    }
    #pragma unroll
    for (int fb = 0; fb < 2; ++fb) {
      const unsigned short* p = &sB[(wn * 32 + fb * 16 + mr) * LDT];
      vbf8 lo = *(const vbf8*)(p + 8 * lh);
      vbf8 hi = *(const vbf8*)(p + 16 + 8 * lh);
      #pragma unroll
      for (int j = 0; j < 8; ++j) { bfr[fb][j] = lo[j]; bfr[fb][8 + j] = hi[j]; }
    }
    #pragma unroll
    for (int fa = 0; fa < 4; ++fa)
      #pragma unroll
      for (int fb = 0; fb < 2; ++fb)
        acc[fa][fb] = __builtin_amdgcn_wmma_f32_16x16x32_bf16(
            false, af[fa], false, bfr[fb], (short)0, acc[fa][fb], false, false);
    __syncthreads();
  }

  // epilogue: lane -> (N = mr, M = r + 8*lh) within each 16x16 fragment
  #pragma unroll
  for (int fa = 0; fa < 4; ++fa)
    #pragma unroll
    for (int fb = 0; fb < 2; ++fb) {
      int col = blockN + wn * 32 + fb * 16 + mr;
      #pragma unroll
      for (int r = 0; r < 8; ++r) {
        int row = blockM + wm * 64 + fa * 16 + 8 * lh + r;
        float v = acc[fa][fb][r];
        if (EPI == 0) {
          ((unsigned short*)Cout)[(size_t)row * N + col] = f2bf(v);
        } else if (EPI == 1) {
          float t = v + bias[col];
          ((float*)Cout)[(size_t)row * N + col] = 1.0f + __expf(t); // exp(softplus(t))
        } else {
          ((float*)Cout)[(size_t)row * N + col] = v;
        }
      }
    }
}

// ---------- causal depthwise conv1d (reads xh = xz[:, :DI] in bf16) ----------
__global__ void conv_kernel(const unsigned short* __restrict__ xz,
                            const float* __restrict__ cw,
                            const float* __restrict__ cb,
                            unsigned short* __restrict__ xc) {
  int idx = blockIdx.x * blockDim.x + threadIdx.x;
  if (idx >= NT * DI) return;
  int d  = idx & (DI - 1);
  int bt = idx >> 11;            // b*SQ + t
  int t  = bt & (SQ - 1);
  float acc = cb[d];
  #pragma unroll
  for (int j = 0; j < DC; ++j) {
    int tt = t - (DC - 1) + j;
    if (tt >= 0)
      acc += cw[d * DC + j] * bf2f(xz[(size_t)(bt - (DC - 1) + j) * (2 * DI) + d]);
  }
  xc[(size_t)bt * DI + d] = f2bf(acc);
}

// ---------- pack AB[d][0:16] = -exp(A_log), AB[d][16:32] = B ----------
__global__ void prep_ab_kernel(const float* __restrict__ A_log,
                               const float* __restrict__ Bm,
                               float* __restrict__ AB) {
  int i = blockIdx.x * blockDim.x + threadIdx.x;
  if (i >= DI * DS) return;
  int d = i >> 4, e = i & 15;
  AB[d * 32 + e]      = -__expf(A_log[i]);
  AB[d * 32 + 16 + e] = Bm[i];
}

// ---------- EF[t][e32] = sum_d ED[t][d] * AB[d][e32]  (E | F concatenated) ----------
__global__ __launch_bounds__(256)
void ef_kernel(const float* __restrict__ ED, const float* __restrict__ AB,
               float* __restrict__ EF) {
  __shared__ float red[256];
  int t = blockIdx.x;
  int tid = threadIdx.x;
  int e = tid & 31, chunk = tid >> 5;       // 8 chunks of 256 d-values
  const float* row = ED + (size_t)t * DI;
  float acc = 0.f;
  int d0 = chunk * 256;
  for (int d = d0; d < d0 + 256; ++d) acc += row[d] * AB[d * 32 + e];
  red[tid] = acc;
  __syncthreads();
  if (chunk == 0) {
    float s = acc;
    #pragma unroll
    for (int c = 1; c < 8; ++c) s += red[e + 32 * c];
    EF[(size_t)t * 32 + e] = s;
  }
}

// ---------- Blelloch up/down sweep, exact replication, h in 128KB LDS ----------
__global__ __launch_bounds__(1024)
void scan_kernel(const float* __restrict__ EF, float* __restrict__ hout) {
  extern __shared__ float hsh[];            // SQ*DS floats = 128KB
  int b = blockIdx.x, tid = threadIdx.x;
  const float* ef = EF + (size_t)b * SQ * 32;
  for (int i = tid; i < SQ * DS; i += 1024) hsh[i] = 0.f;
  __syncthreads();
  // up sweep: i = 0..L-1
  for (int lev = 0; lev < 11; ++lev) {
    int blk = 1 << lev, stride = blk << 1;
    int total = (SQ / stride) * DS;
    for (int idx = tid; idx < total; idx += 1024) {
      int nn = idx >> 4, e = idx & 15;
      int pos = nn * stride;
      float hv = hsh[pos * DS + e] + ef[pos * 32 + e];        // h[::2blk] += E
      hsh[pos * DS + e] = hv;
      if (lev < 10) {
        int pos2 = pos + blk;
        hsh[pos2 * DS + e] = hv + ef[pos2 * 32 + 16 + e];     // set: updated h + F
      }
    }
    __syncthreads();
  }
  // down sweep: i = L-2..0
  for (int lev = 9; lev >= 0; --lev) {
    int blk = 1 << lev, stride = blk << 1;
    int total = (SQ / stride) * DS;
    for (int idx = tid; idx < total; idx += 1024) {
      int nn = idx >> 4, e = idx & 15;
      int pos = nn * stride, pos2 = pos + blk;
      hsh[pos2 * DS + e] += ef[pos * 32 + e] * hsh[pos * DS + e];
    }
    __syncthreads();
  }
  for (int i = tid; i < SQ * DS; i += 1024)
    hout[(size_t)b * SQ * DS + i] = hsh[i];
}

// ---------- y = h*C^T + D*xconv, * silu(z), -> bf16 ----------
__global__ void yact_kernel(const float* __restrict__ hv,
                            const float* __restrict__ Cm,
                            const float* __restrict__ Dv,
                            const unsigned short* __restrict__ xc,
                            const unsigned short* __restrict__ xz,
                            unsigned short* __restrict__ ya) {
  int idx = blockIdx.x * blockDim.x + threadIdx.x;
  if (idx >= NT * DI) return;
  int d  = idx & (DI - 1);
  int bt = idx >> 11;
  const float* h = hv + (size_t)bt * DS;
  float y = 0.f;
  #pragma unroll
  for (int e = 0; e < DS; ++e) y += h[e] * Cm[d * DS + e];
  y += Dv[d] * bf2f(xc[idx]);
  float z = bf2f(xz[(size_t)bt * (2 * DI) + DI + d]);
  float s = z / (1.f + __expf(-z));
  ya[idx] = f2bf(y * s);
}

// ---------- host ----------
static inline void* wsal(char*& p, size_t bytes) {
  void* r = (void*)p;
  p += (bytes + 255) & ~(size_t)255;
  return r;
}

extern "C" void kernel_launch(void* const* d_in, const int* in_sizes, int n_in,
                              void* d_out, int out_size, void* d_ws, size_t ws_size,
                              hipStream_t stream) {
  (void)in_sizes; (void)n_in; (void)out_size; (void)ws_size;
  const float* x         = (const float*)d_in[0];
  const float* in_proj_w = (const float*)d_in[1];
  const float* conv_w    = (const float*)d_in[2];
  const float* conv_b    = (const float*)d_in[3];
  const float* dt_proj_w = (const float*)d_in[4];
  const float* dt_proj_b = (const float*)d_in[5];
  const float* A_log     = (const float*)d_in[6];
  const float* Bmat      = (const float*)d_in[7];
  const float* Cmat      = (const float*)d_in[8];
  const float* Dvec      = (const float*)d_in[9];
  const float* out_proj_w= (const float*)d_in[10];

  char* p = (char*)d_ws;
  unsigned short* xbf    = (unsigned short*)wsal(p, (size_t)NT * DM * 2);      // 16MB
  unsigned short* w1bf   = (unsigned short*)wsal(p, (size_t)2 * DI * DM * 2);  // 8MB
  unsigned short* wdtbf  = (unsigned short*)wsal(p, (size_t)DI * DI * 2);      // 8MB
  unsigned short* woutbf = (unsigned short*)wsal(p, (size_t)DM * DI * 2);      // 4MB
  unsigned short* xzbf   = (unsigned short*)wsal(p, (size_t)NT * 2 * DI * 2);  // 64MB
  unsigned short* xcbf   = (unsigned short*)wsal(p, (size_t)NT * DI * 2);      // 32MB
  float*          ED     = (float*)wsal(p, (size_t)NT * DI * 4);               // 64MB
  float*          AB     = (float*)wsal(p, (size_t)DI * 32 * 4);
  float*          EF     = (float*)wsal(p, (size_t)NT * 32 * 4);
  float*          hbuf   = (float*)wsal(p, (size_t)NT * DS * 4);
  unsigned short* yabf   = (unsigned short*)wsal(p, (size_t)NT * DI * 2);      // 32MB

  // 1) fp32 -> bf16 conversions
  {
    int n;
    n = NT * DM;      cvt_bf16_kernel<<<(n + 255) / 256, 256, 0, stream>>>(x, xbf, n);
    n = 2 * DI * DM;  cvt_bf16_kernel<<<(n + 255) / 256, 256, 0, stream>>>(in_proj_w, w1bf, n);
    n = DI * DI;      cvt_bf16_kernel<<<(n + 255) / 256, 256, 0, stream>>>(dt_proj_w, wdtbf, n);
    n = DM * DI;      cvt_bf16_kernel<<<(n + 255) / 256, 256, 0, stream>>>(out_proj_w, woutbf, n);
  }

  // 2) in_proj: xz[NT, 2*DI] = x @ in_proj_w^T   (bf16 out)
  {
    dim3 grid((2 * DI) / 128, NT / 128);
    wmma_gemm_kernel<0><<<grid, 256, 0, stream>>>(xbf, w1bf, xzbf, nullptr, NT, 2 * DI, DM);
  }

  // 3) causal depthwise conv
  {
    int n = NT * DI;
    conv_kernel<<<(n + 255) / 256, 256, 0, stream>>>(xzbf, conv_w, conv_b, xcbf);
  }

  // 4) dt_proj + softplus + exp fused epilogue: ED = 1 + exp(xc @ Wdt^T + b)
  {
    dim3 grid(DI / 128, NT / 128);
    wmma_gemm_kernel<1><<<grid, 256, 0, stream>>>(xcbf, wdtbf, ED, dt_proj_b, NT, DI, DI);
  }

  // 5) AB pack, then E|F = ED @ [ -exp(A_log) | B ]
  {
    int n = DI * DS;
    prep_ab_kernel<<<(n + 255) / 256, 256, 0, stream>>>(A_log, Bmat, AB);
    ef_kernel<<<NT, 256, 0, stream>>>(ED, AB, EF);
  }

  // 6) Blelloch scan per batch (128KB dynamic LDS)
  scan_kernel<<<NB, 1024, SQ * DS * sizeof(float), stream>>>(EF, hbuf);

  // 7) y = (h @ C^T + D*xc) * silu(z) -> bf16
  {
    int n = NT * DI;
    yact_kernel<<<(n + 255) / 256, 256, 0, stream>>>(hbuf, Cmat, Dvec, xcbf, xzbf, yabf);
  }

  // 8) out = y @ out_proj_w^T  (fp32 out)
  {
    dim3 grid(DM / 128, NT / 128);
    wmma_gemm_kernel<2><<<grid, 256, 0, stream>>>(yabf, woutbf, d_out, nullptr, NT, DM, DI);
  }
}
